// AttentionLayer_30958124270376
// MI455X (gfx1250) — matrix operand order repaired and checked by reading, not compiled, not run on hardware
//
#include <hip/hip_runtime.h>

// ---------------------------------------------------------------------------
// GAT layer for MI455X (gfx1250, wave32, WMMA + async global->LDS).
//
// Algebra: with scores = adj ? s_src[i]+s_dst[j] : 0,
//   out[h,i,:] = ( e^{s_src}·(adj @ (e^{s_dst}⊙h)) + T - adj@h )
//              / ( e^{s_src}·(adj @ e^{s_dst})     + N - deg  )
// -> everything is ONE GEMM  adj(6144x6144) @ V(6144x576 padded),
// V = per-head [e^{s_dst}·h | h] x4 | e^{s_dst}(4) | ones | pad.
// adj is 0/1 -> exact in f16 -> v_wmma_f32_16x16x32_f16, f32 accumulate.
// HBM floor: one 151MB int32 adj pass ≈ 6.5us @ 23.3TB/s; V (6.75MB f16)
// stays L2-resident. K-loop is 2-stage pipelined: V tile moves via
// GLOBAL_LOAD_ASYNC_TO_LDS_B128 (ASYNCcnt) into the back LDS buffer while
// WMMAs run on the front buffer; adj tile is fetched to VGPRs early and
// converted/stored to LDS after the WMMAs. Wave tile is 32x144 (2 A-frags,
// 9 B-frags, 18 WMMAs) so each B fragment feeds TWO WMMAs -> half the
// ds_load/wait traffic per matrix op vs a 16x288 tile.
// ---------------------------------------------------------------------------

typedef _Float16 v8h  __attribute__((ext_vector_type(8)));
typedef _Float16 v16h __attribute__((ext_vector_type(16)));
typedef float    v8f  __attribute__((ext_vector_type(8)));

#define GN   6144   // N
#define INF  256    // IN_F
#define NH   4      // heads
#define DH   64
#define OUTF 256    // H*DH
#define KV   576    // padded V columns (36 tiles of 16)

// Wave-relative LDS byte address of a generic pointer to __shared__ memory
// (flat LDS-aperture addresses carry the LDS offset in the low 32 bits).
__device__ __forceinline__ unsigned lds_off(const void* p) {
    return (unsigned)(unsigned long long)p;
}

// ---------------------------------------------------------------------------
// Kernel 1: h = x@W + Wb ; s_src(+Ab) ; e^{s_dst} ; build V^T (f16, [col][n]
// so GEMM B-fragments are K-contiguous) ; T[h,d] = sum_n h via f32 atomics.
// ---------------------------------------------------------------------------
__global__ __launch_bounds__(256) void gat_prolog(
    const float* __restrict__ x,  const float* __restrict__ W,
    const float* __restrict__ Wb, const float* __restrict__ A,
    const float* __restrict__ Ab,
    _Float16* __restrict__ vt, float* __restrict__ ssrc, float* __restrict__ Tbuf)
{
    __shared__ float xs[8][INF];
    __shared__ float red[2][8][256];
    __shared__ float edst[8][NH];

    const int t  = threadIdx.x;
    const int n0 = blockIdx.x * 8;

    #pragma unroll
    for (int r = 0; r < 8; ++r) xs[r][t] = x[(size_t)(n0 + r) * INF + t];
    __syncthreads();

    const int h = t >> 6, d = t & 63;
    float acc[8] = {0.f,0.f,0.f,0.f,0.f,0.f,0.f,0.f};
    const float* wp = W + (size_t)h * INF * DH + d;
    for (int i = 0; i < INF; ++i) {
        float wv = wp[(size_t)i * DH];          // coalesced over lanes (d)
        #pragma unroll
        for (int r = 0; r < 8; ++r) acc[r] += xs[r][i] * wv;
    }
    const float wb   = Wb[h * DH + d];
    const float asrc = A[h * 2 * DH + d];
    const float adst = A[h * 2 * DH + DH + d];

    float hv[8];
    #pragma unroll
    for (int r = 0; r < 8; ++r) hv[r] = acc[r] + wb;
    #pragma unroll
    for (int r = 0; r < 8; ++r) { red[0][r][t] = hv[r] * asrc; red[1][r][t] = hv[r] * adst; }
    __syncthreads();

    if (t < 64) {   // 8 rows x 4 heads x {src,dst} = 64 reductions of 64
        int r = t >> 3, hh = (t >> 1) & 3, kind = t & 1;
        float s = 0.f;
        for (int j = 0; j < 64; ++j) s += red[kind][r][hh * 64 + j];
        if (kind == 0) ssrc[hh * GN + n0 + r] = s + Ab[hh];
        else           edst[r][hh] = __expf(s);
    }
    __syncthreads();

    float tsum = 0.f;
    #pragma unroll
    for (int r = 0; r < 8; ++r) {
        const int n = n0 + r;
        const float e = edst[r][h];
        vt[(size_t)(h * DH + d) * GN + n]       = (_Float16)(e * hv[r]); // E block
        vt[(size_t)(256 + h * DH + d) * GN + n] = (_Float16)hv[r];      // M block
        tsum += hv[r];
    }
    atomicAdd(&Tbuf[t], tsum);                                          // T[h,d]

    if (t < NH) {
        #pragma unroll
        for (int r = 0; r < 8; ++r)
            vt[(size_t)(512 + t) * GN + n0 + r] = (_Float16)edst[r][t]; // S cols
    }
    if (t == NH) {
        #pragma unroll
        for (int r = 0; r < 8; ++r)
            vt[(size_t)516 * GN + n0 + r] = (_Float16)1.0f;             // deg col
    }
    for (int col = 517 + t; col < KV; col += 256) {                     // pad
        #pragma unroll
        for (int r = 0; r < 8; ++r) vt[(size_t)col * GN + n0 + r] = (_Float16)0.f;
    }
}

// ---------------------------------------------------------------------------
// Kernel 2: C(6144x576) = f16(adj) @ V, f32 accumulate, K-step 32.
// Block: 256 thr = 8 waves = 2 row-groups x 4 col-groups -> 64 rows x 576 cols.
// Wave: 32 rows x 144 cols = 2 A-frags x 9 B-frags = 18 WMMAs / K-step,
// 18 x v8f accumulators. Each B fragment feeds both A fragments.
// Double-buffered LDS; V tile via async global->LDS (ASYNCcnt).
// LDS rows padded to 40 halfs (80B) to break the 64B-stride bank conflict.
// ---------------------------------------------------------------------------
__global__ __launch_bounds__(256) void gat_gemm(
    const int* __restrict__ adj, const _Float16* __restrict__ vt,
    float* __restrict__ C)
{
    __shared__ _Float16 lds_a[2][64][40];   // adj tiles: 64 rows x 32 K (f16)
    __shared__ _Float16 lds_b[2][KV][40];   // V tiles:  576 cols x 32 K (f16)

    const int t    = threadIdx.x;
    const int lane = t & 31;
    const int w    = t >> 5;
    const int rg   = w & 1;              // row group 0..1 (32 rows each)
    const int cg   = w >> 1;             // col group 0..3 (144 cols each)
    const int R    = blockIdx.x * 64;

    const v8f vz = {0.f,0.f,0.f,0.f,0.f,0.f,0.f,0.f};
    v8f acc0[9], acc1[9];
    #pragma unroll
    for (int i = 0; i < 9; ++i) { acc0[i] = vz; acc1[i] = vz; }

    const int arow = t >> 2;             // 0..63
    const int ac   = (t & 3) * 8;        // 0,8,16,24
    const int rl0  = rg * 32 + (lane & 15);       // A frag 0 rows
    const int rl1  = rl0 + 16;                    // A frag 1 rows
    const int ko   = (lane >> 4) * 8;    // A frag: lanes 0-15 K0-7/16-23, 16-31 K8-15/24-31
    const int kb   = (lane >> 4) * 16;   // unused; B frags are read at ko, same K order as A

    // ---- staging helpers -------------------------------------------------
    auto issueB = [&](int buf, int kk) {           // async V tile -> LDS
        #pragma unroll
        for (int cc = 0; cc < 3; ++cc) {
            int col = cc * 256 + t;
            if (col < KV) {
                const _Float16* g = vt + (size_t)col * GN + kk;
                _Float16*       l = &lds_b[buf][col][0];
                #pragma unroll
                for (int j = 0; j < 4; ++j) {      // 4 x 16B per column
                    asm volatile(
                        "global_load_async_to_lds_b128 %0, %1, off"
                        :: "v"(lds_off(l + j * 8)), "v"(g + j * 8)
                        : "memory");
                }
            }
        }
    };
    auto loadA = [&](int kk, int4& a0, int4& a1) { // adj int32 -> VGPRs
        const int4* ap = (const int4*)(adj + (size_t)(R + arow) * GN + kk + ac);
        a0 = ap[0]; a1 = ap[1];
    };
    auto storeA = [&](int buf, int4 a0, int4 a1) { // int -> f16 (0/1 exact)
        v8h af = { (_Float16)a0.x, (_Float16)a0.y, (_Float16)a0.z, (_Float16)a0.w,
                   (_Float16)a1.x, (_Float16)a1.y, (_Float16)a1.z, (_Float16)a1.w };
        *(v8h*)&lds_a[buf][arow][ac] = af;
    };

    // ---- prologue: fill buffer 0 for k=0 ---------------------------------
    {
        issueB(0, 0);
        int4 a0, a1;
        loadA(0, a0, a1);
        storeA(0, a0, a1);
        asm volatile("s_wait_asynccnt 0x0" ::: "memory");
        __syncthreads();
    }

    // ---- pipelined K loop ------------------------------------------------
    int cb = 0;
    for (int k0 = 0; k0 < GN; k0 += 32) {
        const int  nb      = cb ^ 1;
        const bool hasNext = (k0 + 32) < GN;

        int4 a0, a1;
        if (hasNext) {                       // stage next tile (overlaps WMMAs)
            issueB(nb, k0 + 32);
            loadA(k0 + 32, a0, a1);
        }

        // Two A fragments (16x32 f16 each, documented wave32 striping)
        v8h a0lo = *(const v8h*)&lds_a[cb][rl0][ko];
        v8h a0hi = *(const v8h*)&lds_a[cb][rl0][16 + ko];
        v8h a1lo = *(const v8h*)&lds_a[cb][rl1][ko];
        v8h a1hi = *(const v8h*)&lds_a[cb][rl1][16 + ko];
        v16h av0 = __builtin_shufflevector(a0lo, a0hi,
                     0,1,2,3,4,5,6,7,8,9,10,11,12,13,14,15);
        v16h av1 = __builtin_shufflevector(a1lo, a1hi,
                     0,1,2,3,4,5,6,7,8,9,10,11,12,13,14,15);

        // 9 column tiles; each B fragment drives both A fragments
        const _Float16* bbase = &lds_b[cb][cg * 144 + (lane & 15)][ko];
        v8h blo = *(const v8h*)(bbase);
        v8h bhi = *(const v8h*)(bbase + 16);
        #pragma unroll
        for (int tt = 0; tt < 9; ++tt) {
            v8h nlo, nhi;
            if (tt < 8) {                    // prefetch tile tt+1 from LDS
                const _Float16* nb_ = bbase + (tt + 1) * 16 * 40;
                nlo = *(const v8h*)(nb_);
                nhi = *(const v8h*)(nb_ + 16);
            }
            v16h bv = __builtin_shufflevector(blo, bhi,
                        0,1,2,3,4,5,6,7,8,9,10,11,12,13,14,15);
            acc0[tt] = __builtin_amdgcn_wmma_f32_16x16x32_f16(
                           false, av0, false, bv, (short)0, acc0[tt], false, false);
            acc1[tt] = __builtin_amdgcn_wmma_f32_16x16x32_f16(
                           false, av1, false, bv, (short)0, acc1[tt], false, false);
            if (tt < 8) { blo = nlo; bhi = nhi; }
        }

        if (hasNext) storeA(nb, a0, a1);     // adj convert -> back buffer
        asm volatile("s_wait_asynccnt 0x0" ::: "memory");
        __syncthreads();                     // waits dscnt too
        cb = nb;
    }

    // ---- writeback: lane L holds (M = v + 8*(L>>4), N = L&15) ------------
    #pragma unroll
    for (int tt = 0; tt < 9; ++tt) {
        #pragma unroll
        for (int v = 0; v < 8; ++v) {
            int colg = cg * 144 + tt * 16 + (lane & 15);
            int rowg = R + rg * 32 + (lane >> 4) * 8 + v;
            C[(size_t)rowg * KV + colg]        = acc0[tt][v];
            C[(size_t)(rowg + 16) * KV + colg] = acc1[tt][v];
        }
    }
}

// ---------------------------------------------------------------------------
// Kernel 3: out[n, h*64+d] = (e^{s_src}·E + T - M) / (e^{s_src}·S + N - deg)
// ---------------------------------------------------------------------------
__global__ __launch_bounds__(256) void gat_epilog(
    const float* __restrict__ C, const float* __restrict__ ssrc,
    const float* __restrict__ Tbuf, float* __restrict__ out)
{
    const int n = blockIdx.x;
    const int c = threadIdx.x;
    const int h = c >> 6;
    const float* cr = C + (size_t)n * KV;
    const float es  = __expf(ssrc[h * GN + n]);
    const float E   = cr[c];
    const float M   = cr[256 + c];
    const float S   = cr[512 + h];
    const float deg = cr[516];
    const float Z   = es * S + ((float)GN - deg);
    out[(size_t)n * OUTF + c] = (es * E + Tbuf[c] - M) / Z;
}

// ---------------------------------------------------------------------------
extern "C" void kernel_launch(void* const* d_in, const int* in_sizes, int n_in,
                              void* d_out, int out_size, void* d_ws, size_t ws_size,
                              hipStream_t stream)
{
    const float* x   = (const float*)d_in[0];
    const int*   adj = (const int*)  d_in[1];
    const float* W   = (const float*)d_in[2];
    const float* Wb  = (const float*)d_in[3];
    const float* A   = (const float*)d_in[4];
    const float* Ab  = (const float*)d_in[5];
    float* out = (float*)d_out;

    // workspace carve-up (~21.3 MB)
    char* ws = (char*)d_ws;
    _Float16* vt  = (_Float16*)ws;                                  // KV*GN*2
    float*    C   = (float*)(ws + (size_t)KV * GN * 2);             // GN*KV*4
    float*    ssrc= (float*)(ws + (size_t)KV * GN * 2 + (size_t)GN * KV * 4);
    float*    Tbuf= ssrc + (size_t)NH * GN;

    hipMemsetAsync(Tbuf, 0, OUTF * sizeof(float), stream);
    gat_prolog<<<GN / 8, 256, 0, stream>>>(x, W, Wb, A, Ab, vt, ssrc, Tbuf);
    gat_gemm  <<<GN / 64, 256, 0, stream>>>(adj, vt, C);
    gat_epilog<<<GN, 256, 0, stream>>>(C, ssrc, Tbuf, out);
    (void)in_sizes; (void)n_in; (void)out_size; (void)ws_size;
}
